// RoPEMultiheadAttention_78280073937010
// MI455X (gfx1250) — matrix operand order repaired and checked by reading, not compiled, not run on hardware
//
#include <hip/hip_runtime.h>

typedef _Float16 f16;
typedef __attribute__((ext_vector_type(4)))  float    v4f;
typedef __attribute__((ext_vector_type(8)))  float    v8f;
typedef __attribute__((ext_vector_type(4)))  _Float16 v4h;
typedef __attribute__((ext_vector_type(8)))  _Float16 v8h;
typedef __attribute__((ext_vector_type(16))) _Float16 v16h;

#define SEQ 1024
#define DIM 1024
#define NH  16
#define HD  64
#define NB  4

// ---------------------------------------------------------------------------
// WMMA helpers (CDNA5 16x16x32 f16 -> f32), wave32 layouts per CDNA5 ISA:
//   A (16x32, MxK): lane<16 -> M=lane, K in {k0..k0+7, k0+16..k0+23}
//                   lane>=16 -> M=lane-16, K in {k0+8..k0+15, k0+24..k0+31}
//   B (32x16, KxN): lane<16 -> N=lane, K = same set as A lane<16 ; lane>=16 -> N=lane-16, K = same set as A lane>=16
//   C/D (16x16 f32): vgpr j, lane l -> M = j + 8*(l>>4), N = l&15
// ---------------------------------------------------------------------------
static __device__ __forceinline__ v8f wmma16(v16h a, v16h b, v8f c) {
  return __builtin_amdgcn_wmma_f32_16x16x32_f16(false, a, false, b, (short)0, c,
                                                false, false);
}

static __device__ __forceinline__ v16h load_a16(const f16* base, int row0,
                                                int stride, int k0, int r, int half) {
  const f16* p = base + (size_t)(row0 + r) * stride + k0 + half * 8;
  v8h lo = *(const v8h*)p;          // K: half0 -> 0..7,   half1 -> 8..15
  v8h hi = *(const v8h*)(p + 16);   // K: half0 -> 16..23, half1 -> 24..31
  v16h a;
#pragma unroll
  for (int i = 0; i < 8; ++i) { a[i] = lo[i]; a[i + 8] = hi[i]; }
  return a;
}

static __device__ __forceinline__ v16h load_b16(const f16* base, int col0,
                                                int stride, int k0, int r, int half) {
  const f16* p = base + (size_t)(col0 + r) * stride + k0 + half * 8;
  v8h lo = *(const v8h*)p;          // K: half0 -> 0..7,   half1 -> 8..15
  v8h hi = *(const v8h*)(p + 16);   // K: half0 -> 16..23, half1 -> 24..31
  v16h b;
#pragma unroll
  for (int i = 0; i < 8; ++i) { b[i] = lo[i]; b[i + 8] = hi[i]; }
  return b;
}

// ---------------------------------------------------------------------------
// fp32 -> f16 conversion prepass (memory bound, vectorized x4)
// ---------------------------------------------------------------------------
__global__ void cvt_kernel(const float* __restrict__ src, f16* __restrict__ dst, int n) {
  int i = (blockIdx.x * blockDim.x + threadIdx.x) * 4;
  if (i < n) {
    v4f v = *(const v4f*)(src + i);
    v4h o;
    o[0] = (f16)v[0]; o[1] = (f16)v[1]; o[2] = (f16)v[2]; o[3] = (f16)v[3];
    *(v4h*)(dst + i) = o;
  }
}

// ---------------------------------------------------------------------------
// Stage 1: Q/K/V projections (Y = X @ W^T) with fused RoPE on Q,K.
// grid = (32, 16, 3); block = 256 (8 waves). Each wave: one 16x64 output tile
// (register-blocked: A loaded once per K-step, reused across 4 B tiles ->
//  4 WMMAs per 10 b128 loads instead of 4 per 16).
//   z==0: Q (RoPE + 1/sqrt(64) scale) -> qr[bh][m][od]
//   z==1: K (RoPE)                    -> kr[bh][m][od]
//   z==2: V (transposed)              -> vt[bh][od][m]
// The 64-wide tile spans exactly one head: h is wave-uniform, od = t*16 + r.
// ---------------------------------------------------------------------------
__global__ void qkv_rope_kernel(const f16* __restrict__ xb, const f16* __restrict__ wb,
                                const float* __restrict__ R,
                                f16* __restrict__ qr, f16* __restrict__ kr,
                                f16* __restrict__ vt) {
  const int lane = threadIdx.x & 31;
  const int wave = threadIdx.x >> 5;
  const int half = lane >> 4;
  const int r    = lane & 15;
  const int nbase = (blockIdx.x * 8 + wave) * 16;   // row in [0, 4096)
  const int obase = blockIdx.y * 64;                // head-aligned col base
  const int which = blockIdx.z;
  const f16* w = wb + (size_t)which * DIM * DIM;

  v8f c0 = {0.f,0.f,0.f,0.f,0.f,0.f,0.f,0.f};
  v8f c1 = c0, c2 = c0, c3 = c0;
#pragma unroll 2
  for (int k0 = 0; k0 < DIM; k0 += 32) {
    v16h a  = load_a16(xb, nbase, DIM, k0, r, half);
    v16h b0 = load_b16(w, obase +  0, DIM, k0, r, half);
    v16h b1 = load_b16(w, obase + 16, DIM, k0, r, half);
    v16h b2 = load_b16(w, obase + 32, DIM, k0, r, half);
    v16h b3 = load_b16(w, obase + 48, DIM, k0, r, half);
    c0 = wmma16(a, b0, c0);
    c1 = wmma16(a, b1, c1);
    c2 = wmma16(a, b2, c2);
    c3 = wmma16(a, b3, c3);
  }
  v8f acc[4] = {c0, c1, c2, c3};

  const int h = obase >> 6;                         // wave-uniform head index
  if (which < 2) {
    f16* dst = (which == 0) ? qr : kr;
    const float qs = (which == 0) ? 0.125f : 1.0f;  // fold 1/sqrt(HD) into Q
#pragma unroll
    for (int t = 0; t < 4; ++t) {
#pragma unroll
      for (int j = 0; j < 8; ++j) {
        const int n = nbase + j + 8 * half;         // global row
        const int b = n >> 10, m = n & 1023;
        const int od = t * 16 + r;                  // dim within head
        const float self = acc[t][j];
        // RoPE pair (od, od^1) lives in the adjacent lane (N differs in bit 0)
        const float part = __shfl_xor(self, 1, 32);
        const float* Rr = R + ((size_t)m * HD + od) * HD;
        const float val = (Rr[od] * self + Rr[od ^ 1] * part) * qs;
        dst[((size_t)((b * NH + h) * SEQ + m)) * HD + od] = (f16)val;
      }
    }
  } else {
#pragma unroll
    for (int t = 0; t < 4; ++t) {
#pragma unroll
      for (int j = 0; j < 8; ++j) {
        const int n = nbase + j + 8 * half;
        const int b = n >> 10, m = n & 1023;
        const int od = t * 16 + r;
        vt[((size_t)((b * NH + h) * HD + od)) * SEQ + m] = (f16)acc[t][j];
      }
    }
  }
}

// ---------------------------------------------------------------------------
// Stage 2: flash attention per (b,h). grid = (8, 64); block = 256 (8 waves).
// Wave owns 16 query rows; loops 1024 keys in blocks of 32 with online softmax.
// P tile is re-layouted C->A through a wave-private LDS tile.
// ---------------------------------------------------------------------------
__global__ void attn_kernel(const f16* __restrict__ qr, const f16* __restrict__ kr,
                            const f16* __restrict__ vt, float* __restrict__ out) {
  __shared__ f16 lds[8 * 16 * 32] __attribute__((aligned(32)));
  const int lane = threadIdx.x & 31;
  const int wave = threadIdx.x >> 5;
  const int half = lane >> 4;
  const int r    = lane & 15;
  const int bh    = blockIdx.y;
  const int qbase = blockIdx.x * 128 + wave * 16;
  const f16* qp = qr + (size_t)bh * SEQ * HD;
  const f16* kp = kr + (size_t)bh * SEQ * HD;
  const f16* vp = vt + (size_t)bh * HD * SEQ;
  f16* myl = lds + wave * (16 * 32);

  // Q tile (16 rows x 64 dims) stays in registers the whole loop
  const v16h aq0 = load_a16(qp, qbase, HD, 0,  r, half);
  const v16h aq1 = load_a16(qp, qbase, HD, 32, r, half);

  float mrun[8], lrun[8];
  v8f acc0 = {0.f,0.f,0.f,0.f,0.f,0.f,0.f,0.f};
  v8f acc1 = acc0, acc2 = acc0, acc3 = acc0;
#pragma unroll
  for (int j = 0; j < 8; ++j) { mrun[j] = -3.0e38f; lrun[j] = 0.f; }

  for (int kb = 0; kb < SEQ; kb += 32) {
    // scores: S = (Q/8) . K^T  (K dim = head dim 64, two 32-chunks)
    v16h bk00 = load_b16(kp, kb,      HD, 0,  r, half);
    v16h bk01 = load_b16(kp, kb,      HD, 32, r, half);
    v16h bk10 = load_b16(kp, kb + 16, HD, 0,  r, half);
    v16h bk11 = load_b16(kp, kb + 16, HD, 32, r, half);
    v8f z = {0.f,0.f,0.f,0.f,0.f,0.f,0.f,0.f};
    v8f s0 = wmma16(aq1, bk01, wmma16(aq0, bk00, z));
    v8f s1 = wmma16(aq1, bk11, wmma16(aq0, bk10, z));

    // online softmax: each C row spans one 16-lane half -> xor{1,2,4,8} reduce
#pragma unroll
    for (int j = 0; j < 8; ++j) {
      float t = fmaxf(s0[j], s1[j]);
      t = fmaxf(t, __shfl_xor(t, 1, 32));
      t = fmaxf(t, __shfl_xor(t, 2, 32));
      t = fmaxf(t, __shfl_xor(t, 4, 32));
      t = fmaxf(t, __shfl_xor(t, 8, 32));
      const float mnew  = fmaxf(mrun[j], t);
      const float alpha = __expf(mrun[j] - mnew);
      const float e0 = __expf(s0[j] - mnew);
      const float e1 = __expf(s1[j] - mnew);
      float rs = e0 + e1;
      rs += __shfl_xor(rs, 1, 32);
      rs += __shfl_xor(rs, 2, 32);
      rs += __shfl_xor(rs, 4, 32);
      rs += __shfl_xor(rs, 8, 32);
      lrun[j] = lrun[j] * alpha + rs;
      mrun[j] = mnew;
      acc0[j] *= alpha; acc1[j] *= alpha; acc2[j] *= alpha; acc3[j] *= alpha;
      s0[j] = e0; s1[j] = e1;
    }

    // P: C layout -> A layout via wave-private LDS tile (16 x 32 f16)
#pragma unroll
    for (int j = 0; j < 8; ++j) {
      const int row = j + 8 * half;
      myl[row * 32 + r]      = (f16)s0[j];
      myl[row * 32 + 16 + r] = (f16)s1[j];
    }
    __syncthreads();
    const f16* pr = myl + r * 32 + half * 8;
    v8h plo = *(const v8h*)pr;
    v8h phi = *(const v8h*)(pr + 16);
    v16h pa;
#pragma unroll
    for (int i = 0; i < 8; ++i) { pa[i] = plo[i]; pa[i + 8] = phi[i]; }
    __syncthreads();

    // O += P . V  (V stored transposed: B lanes read contiguous key runs)
    v16h bv0 = load_b16(vp, 0,  SEQ, kb, r, half);
    v16h bv1 = load_b16(vp, 16, SEQ, kb, r, half);
    v16h bv2 = load_b16(vp, 32, SEQ, kb, r, half);
    v16h bv3 = load_b16(vp, 48, SEQ, kb, r, half);
    acc0 = wmma16(pa, bv0, acc0);
    acc1 = wmma16(pa, bv1, acc1);
    acc2 = wmma16(pa, bv2, acc2);
    acc3 = wmma16(pa, bv3, acc3);
  }

  const int b = bh >> 4, h = bh & 15;
#pragma unroll
  for (int j = 0; j < 8; ++j) {
    const int m = qbase + j + 8 * half;
    const float inv = 1.0f / lrun[j];
    const size_t base = ((size_t)(b * SEQ + m)) * DIM + h * HD + r;
    out[base +  0] = acc0[j] * inv;
    out[base + 16] = acc1[j] * inv;
    out[base + 32] = acc2[j] * inv;
    out[base + 48] = acc3[j] * inv;
  }
}

// ---------------------------------------------------------------------------
extern "C" void kernel_launch(void* const* d_in, const int* in_sizes, int n_in,
                              void* d_out, int out_size, void* d_ws, size_t ws_size,
                              hipStream_t stream) {
  (void)in_sizes; (void)n_in; (void)out_size; (void)ws_size;
  const float* x  = (const float*)d_in[0];
  const float* wq = (const float*)d_in[1];
  const float* wk = (const float*)d_in[2];
  const float* wv = (const float*)d_in[3];
  const float* R  = (const float*)d_in[4];

  // workspace: xb(8MB) | wb(6MB) | qr(8MB) | kr(8MB) | vt(8MB) = 38MB f16
  f16* xb = (f16*)d_ws;
  f16* wb = xb + (size_t)NB * SEQ * DIM;
  f16* qr = wb + (size_t)3 * DIM * DIM;
  f16* kr = qr + (size_t)NB * NH * SEQ * HD;
  f16* vt = kr + (size_t)NB * NH * SEQ * HD;

  const int nx = NB * SEQ * DIM;   // 4M
  const int nw = DIM * DIM;        // 1M
  cvt_kernel<<<nx / 1024, 256, 0, stream>>>(x, xb, nx);
  cvt_kernel<<<nw / 1024, 256, 0, stream>>>(wq, wb, nw);
  cvt_kernel<<<nw / 1024, 256, 0, stream>>>(wk, wb + nw, nw);
  cvt_kernel<<<nw / 1024, 256, 0, stream>>>(wv, wb + 2 * nw, nw);

  qkv_rope_kernel<<<dim3(32, 16, 3), 256, 0, stream>>>(xb, wb, R, qr, kr, vt);
  attn_kernel<<<dim3(8, 64), 256, 0, stream>>>(qr, kr, vt, (float*)d_out);
}
